// GATClassifier_54537494724781
// MI455X (gfx1250) — matrix operand order repaired and checked by reading, not compiled, not run on hardware
//
#include <hip/hip_runtime.h>
#include <hip/hip_bf16.h>
#include <cmath>

#define N_NODES 100000
#define N_EDGES 1600000
#define TDIM    768
#define VDIM    2048
#define ADIM    512
#define HC      32
#define NHEAD   4
#define NVN     20000
#define NAN_    20000
#define BATCH   64

typedef __attribute__((ext_vector_type(16))) __bf16 v16bf;
typedef __attribute__((ext_vector_type(8)))  float  v8f;
typedef unsigned short u16;

union ABv { uint4 q[2]; v16bf v; };   // 32 bytes: q[0] -> VGPR0-3 pairs, q[1] -> VGPR4-7

__device__ __forceinline__ u16 f2bf(float f) {
    unsigned int u = __float_as_uint(f);
    u += 0x7FFFu + ((u >> 16) & 1u);           // round-to-nearest-even
    return (u16)(u >> 16);
}

__device__ __forceinline__ void atomicMaxF(float* addr, float value) {
    if (value >= 0.f) atomicMax((int*)addr, __float_as_int(value));
    else              atomicMin((unsigned int*)addr, __float_as_uint(value));
}

// ---------------------------------------------------------------------------
// f32 -> bf16 streaming conversion (4 elements / thread / iter)
// ---------------------------------------------------------------------------
__global__ void cvt_bf16_kernel(const float* __restrict__ in, u16* __restrict__ out, int n) {
    int i = (blockIdx.x * blockDim.x + threadIdx.x) << 2;
    const int stride = (gridDim.x * blockDim.x) << 2;
    for (; i + 3 < n; i += stride) {
        const float4 v = *(const float4*)(in + i);
        ushort4 o;
        o.x = f2bf(v.x); o.y = f2bf(v.y); o.z = f2bf(v.z); o.w = f2bf(v.w);
        *(ushort4*)(out + i) = o;
    }
}

// ---------------------------------------------------------------------------
// Y(nrows,Cout) = Xb(nrows,K)bf16 @ Wb(Cout,K)bf16^T + bias
// One wave = 32x32 output (2x2 WMMA tiles); per K-step: 8x b128 loads, 4 WMMAs.
// bf16 A-layout: lane klo_a in {0,16}; lane's 16 vals = 32 contiguous bytes at 2*klo_a
// bf16 B-layout: lane klo_b in {0,16}; lane's 16 vals = 32 contiguous bytes at 2*klo_b
// ---------------------------------------------------------------------------
__global__ void gemm_bias_wmma_bf16(const u16* __restrict__ Xb, const u16* __restrict__ Wb,
                                    const float* __restrict__ bias, float* __restrict__ Y,
                                    int nrows, int K, int Cout) {
    const int lane   = threadIdx.x & 31;
    const int gwave  = (blockIdx.x * blockDim.x + threadIdx.x) >> 5;
    const int nwaves = (gridDim.x * blockDim.x) >> 5;
    const int ngrp   = Cout >> 5;
    const int tiles  = (nrows >> 5) * ngrp;
    const int l15    = lane & 15;
    const int aoff   = (lane < 16) ? 0 : 16;
    const int boff   = (lane < 16) ? 0 : 16;
    for (int t = gwave; t < tiles; t += nwaves) {
        const int rt = t / ngrp, nt = t - rt * ngrp;
        const u16* a0p = Xb + (size_t)((rt << 5) + l15) * K + aoff;
        const u16* a1p = a0p + (size_t)16 * K;
        const u16* b0p = Wb + (size_t)((nt << 5) + l15) * K + boff;
        const u16* b1p = b0p + (size_t)16 * K;
        v8f acc00 = {0.f,0.f,0.f,0.f,0.f,0.f,0.f,0.f};
        v8f acc01 = acc00, acc10 = acc00, acc11 = acc00;
        for (int k0 = 0; k0 < K; k0 += 32) {
            ABv A0, A1, B0, B1;
            A0.q[0] = *(const uint4*)(a0p + k0);
            A0.q[1] = *(const uint4*)(a0p + k0 + 8);
            A1.q[0] = *(const uint4*)(a1p + k0);
            A1.q[1] = *(const uint4*)(a1p + k0 + 8);
            B0.q[0] = *(const uint4*)(b0p + k0);
            B0.q[1] = *(const uint4*)(b0p + k0 + 8);
            B1.q[0] = *(const uint4*)(b1p + k0);
            B1.q[1] = *(const uint4*)(b1p + k0 + 8);
            acc00 = __builtin_amdgcn_wmma_f32_16x16x32_bf16(false, A0.v, false, B0.v, (short)0, acc00, false, false);
            acc01 = __builtin_amdgcn_wmma_f32_16x16x32_bf16(false, A0.v, false, B1.v, (short)0, acc01, false, false);
            acc10 = __builtin_amdgcn_wmma_f32_16x16x32_bf16(false, A1.v, false, B0.v, (short)0, acc10, false, false);
            acc11 = __builtin_amdgcn_wmma_f32_16x16x32_bf16(false, A1.v, false, B1.v, (short)0, acc11, false, false);
        }
        const int col0 = (nt << 5) + l15;
        const int col1 = col0 + 16;
        const float bv0 = bias[col0];
        const float bv1 = bias[col1];
        const int mb = (rt << 5) + ((lane < 16) ? 0 : 8);
#pragma unroll
        for (int r = 0; r < 8; ++r) {
            const size_t row0 = (size_t)(mb + r) * Cout;
            const size_t row1 = (size_t)(mb + 16 + r) * Cout;
            Y[row0 + col0] = acc00[r] + bv0;
            Y[row0 + col1] = acc01[r] + bv1;
            Y[row1 + col0] = acc10[r] + bv0;
            Y[row1 + col1] = acc11[r] + bv1;
        }
    }
}

__global__ void fill_kernel(float* __restrict__ p, float v, int n) {
    int i = blockIdx.x * blockDim.x + threadIdx.x;
    const int stride = gridDim.x * blockDim.x;
    for (; i < n; i += stride) p[i] = v;
}

// fx[idx[r]][c] = relu(LN(hraw[r])) * mean(x[idx[r]])   -- one wave per row
__global__ void fuse_modal_kernel(float* __restrict__ fx, const float* __restrict__ x,
                                  const float* __restrict__ hraw, const int* __restrict__ idx,
                                  const float* __restrict__ g, const float* __restrict__ beta,
                                  int nrows, int xdim) {
    const int lane = threadIdx.x & 31;
    int wave = (blockIdx.x * blockDim.x + threadIdx.x) >> 5;
    const int nwaves = (gridDim.x * blockDim.x) >> 5;
    for (int row = wave; row < nrows; row += nwaves) {
        float v = hraw[(size_t)row * HC + lane];
        float s = v;
#pragma unroll
        for (int off = 16; off; off >>= 1) s += __shfl_xor(s, off);
        const float mean = s * (1.0f / HC);
        const float d = v - mean;
        float vs = d * d;
#pragma unroll
        for (int off = 16; off; off >>= 1) vs += __shfl_xor(vs, off);
        const float rstd = rsqrtf(vs * (1.0f / HC) + 1e-5f);
        float y = fmaxf(d * rstd * g[lane] + beta[lane], 0.f);
        const int node = idx[row];
        const float* xr = x + (size_t)node * xdim;
        float xs = 0.f;
        for (int j = lane; j < xdim; j += 32) xs += xr[j];
#pragma unroll
        for (int off = 16; off; off >>= 1) xs += __shfl_xor(xs, off);
        fx[(size_t)node * HC + lane] = y * (xs / xdim);
    }
}

// pass 1: e = <leaky_relu(xl[s]+xr[d]), att>, segment-max into mbuf
__global__ void edge_score_kernel(const float* __restrict__ xl, const float* __restrict__ xr,
                                  const float* __restrict__ att,
                                  const int* __restrict__ src, const int* __restrict__ dst,
                                  float* __restrict__ ebuf, float* __restrict__ mbuf,
                                  int nedges, int nnodes, int h) {
    const int D = h * HC;
    int i = blockIdx.x * blockDim.x + threadIdx.x;
    const int stride = gridDim.x * blockDim.x;
    const int total = (nedges + nnodes) * h;
    for (; i < total; i += stride) {
        const int e = i / h, hh = i - e * h;
        int s, d;
        if (e < nedges) { s = src[e]; d = dst[e]; } else { s = e - nedges; d = s; }
        const float4* pl = (const float4*)(xl + (size_t)s * D + hh * HC);
        const float4* pr = (const float4*)(xr + (size_t)d * D + hh * HC);
        const float4* pa = (const float4*)(att + hh * HC);
        float acc = 0.f;
#pragma unroll
        for (int c = 0; c < HC / 4; ++c) {
            const float4 a = pl[c], b = pr[c], t = pa[c];
            float v;
            v = a.x + b.x; v = (v > 0.f) ? v : 0.2f * v; acc += v * t.x;
            v = a.y + b.y; v = (v > 0.f) ? v : 0.2f * v; acc += v * t.y;
            v = a.z + b.z; v = (v > 0.f) ? v : 0.2f * v; acc += v * t.z;
            v = a.w + b.w; v = (v > 0.f) ? v : 0.2f * v; acc += v * t.w;
        }
        ebuf[i] = acc;
        atomicMaxF(&mbuf[(size_t)d * h + hh], acc);
    }
}

// pass 2: w = exp(e - m[d]); segment-sum into sbuf
__global__ void edge_softmax_kernel(const int* __restrict__ dst,
                                    const float* __restrict__ mbuf, float* __restrict__ ebuf,
                                    float* __restrict__ sbuf, int nedges, int nnodes, int h) {
    int i = blockIdx.x * blockDim.x + threadIdx.x;
    const int stride = gridDim.x * blockDim.x;
    const int total = (nedges + nnodes) * h;
    for (; i < total; i += stride) {
        const int e = i / h, hh = i - e * h;
        const int d = (e < nedges) ? dst[e] : (e - nedges);
        const float w = __expf(ebuf[i] - mbuf[(size_t)d * h + hh]);
        ebuf[i] = w;
        atomicAdd(&sbuf[(size_t)d * h + hh], w);
    }
}

// pass 3: acc[d] += (w / s[d]) * xl[s]
__global__ void edge_aggregate_kernel(const float* __restrict__ xl,
                                      const int* __restrict__ src, const int* __restrict__ dst,
                                      const float* __restrict__ ebuf, const float* __restrict__ sbuf,
                                      float* __restrict__ acc, int nedges, int nnodes, int h) {
    const int D = h * HC;
    int i = blockIdx.x * blockDim.x + threadIdx.x;
    const int stride = gridDim.x * blockDim.x;
    const int total = (nedges + nnodes) * h;
    for (; i < total; i += stride) {
        const int e = i / h, hh = i - e * h;
        int s, d;
        if (e < nedges) { s = src[e]; d = dst[e]; } else { s = e - nedges; d = s; }
        const int i2 = i + stride;
        if (i2 < total) {          // prefetch next gathered row (global_prefetch_b8)
            const int e2 = i2 / h;
            const int s2 = (e2 < nedges) ? src[e2] : (e2 - nedges);
            __builtin_prefetch(xl + (size_t)s2 * D, 0, 1);
        }
        const float alpha = ebuf[i] / (sbuf[(size_t)d * h + hh] + 1e-16f);
        const float4* pl = (const float4*)(xl + (size_t)s * D + hh * HC);
        float* po = acc + (size_t)d * D + hh * HC;
#pragma unroll
        for (int c = 0; c < HC / 4; ++c) {
            const float4 a = pl[c];
            atomicAdd(po + 4 * c + 0, alpha * a.x);
            atomicAdd(po + 4 * c + 1, alpha * a.y);
            atomicAdd(po + 4 * c + 2, alpha * a.z);
            atomicAdd(po + 4 * c + 3, alpha * a.w);
        }
    }
}

// out = maybe_elu(LN(acc + bias (+ resid)))  -- wave per row; also emits bf16 copy
__global__ void post_kernel(const float* __restrict__ acc, const float* __restrict__ resid,
                            const float* __restrict__ bias, const float* __restrict__ g,
                            const float* __restrict__ beta, float* __restrict__ out,
                            u16* __restrict__ outbf, int n, int D, int do_elu) {
    const int lane = threadIdx.x & 31;
    int wave = (blockIdx.x * blockDim.x + threadIdx.x) >> 5;
    const int nwaves = (gridDim.x * blockDim.x) >> 5;
    const int per = D >> 5;
    for (int row = wave; row < n; row += nwaves) {
        float vals[4];
        float s = 0.f;
        for (int j = 0; j < per; ++j) {
            const int c = (j << 5) + lane;
            float v = acc[(size_t)row * D + c] + bias[c];
            if (resid) v += resid[(size_t)row * D + c];
            vals[j] = v; s += v;
        }
#pragma unroll
        for (int off = 16; off; off >>= 1) s += __shfl_xor(s, off);
        const float mean = s / (float)D;
        float vs = 0.f;
        for (int j = 0; j < per; ++j) { const float dd = vals[j] - mean; vs += dd * dd; }
#pragma unroll
        for (int off = 16; off; off >>= 1) vs += __shfl_xor(vs, off);
        const float rstd = rsqrtf(vs / (float)D + 1e-5f);
        for (int j = 0; j < per; ++j) {
            const int c = (j << 5) + lane;
            float y = (vals[j] - mean) * rstd * g[c] + beta[c];
            if (do_elu) y = (y > 0.f) ? y : (__expf(y) - 1.f);
            out[(size_t)row * D + c] = y;
            if (outbf) outbf[(size_t)row * D + c] = f2bf(y);
        }
    }
}

__global__ void head_kernel(const float* __restrict__ h4, const int* __restrict__ ptr,
                            const float* __restrict__ Wo, const float* __restrict__ bo,
                            float* __restrict__ out, int bcnt) {
    const int b = blockIdx.x * blockDim.x + threadIdx.x;
    if (b < bcnt) {
        const int node = ptr[b];
        float s = bo[0];
#pragma unroll
        for (int c = 0; c < HC; ++c) s += h4[(size_t)node * HC + c] * Wo[c];
        out[b] = s;
    }
}

static inline int imin(int a, int b) { return a < b ? a : b; }

extern "C" void kernel_launch(void* const* d_in, const int* in_sizes, int n_in,
                              void* d_out, int out_size, void* d_ws, size_t ws_size,
                              hipStream_t stream) {
    const float* x        = (const float*)d_in[0];
    const float* x_vision = (const float*)d_in[1];
    const float* x_audio  = (const float*)d_in[2];
    const int*   src      = (const int*)d_in[3];
    const int*   dst      = (const int*)d_in[4];
    const int*   vidx     = (const int*)d_in[5];
    const int*   aidx     = (const int*)d_in[6];
    const int*   ptr      = (const int*)d_in[7];
    const float* Wt  = (const float*)d_in[8];
    const float* bt  = (const float*)d_in[9];
    const float* Wv  = (const float*)d_in[10];
    const float* bv  = (const float*)d_in[11];
    const float* gv  = (const float*)d_in[12];
    const float* bev = (const float*)d_in[13];
    const float* Wa  = (const float*)d_in[14];
    const float* ba  = (const float*)d_in[15];
    const float* ga  = (const float*)d_in[16];
    const float* bea = (const float*)d_in[17];
    const float *cWl[4], *cbl[4], *cWr[4], *cbr[4], *catt[4], *cbias[4];
    for (int L = 0; L < 4; ++L) {
        const int base = 18 + L * 6;
        cWl[L]   = (const float*)d_in[base + 0];
        cbl[L]   = (const float*)d_in[base + 1];
        cWr[L]   = (const float*)d_in[base + 2];
        cbr[L]   = (const float*)d_in[base + 3];
        catt[L]  = (const float*)d_in[base + 4];
        cbias[L] = (const float*)d_in[base + 5];
    }
    const float *ng[4], *nb[4];
    for (int L = 0; L < 4; ++L) {
        ng[L] = (const float*)d_in[42 + L * 2];
        nb[L] = (const float*)d_in[43 + L * 2];
    }
    const float* Wo = (const float*)d_in[50];
    const float* bo = (const float*)d_in[51];

    // ---- workspace carve-up ----
    float* w = (float*)d_ws;
    float* fx   = w; w += (size_t)N_NODES * HC;     // L1 input (f32); reused as h4
    float* hA   = w; w += (size_t)N_NODES * 128;    // h1 / h3; tmpv/tmpa pre-layers
    float* hB   = w; w += (size_t)N_NODES * 128;    // h2
    float* xl   = w; w += (size_t)N_NODES * 128;
    float* xr   = w; w += (size_t)N_NODES * 128;
    float* accb = w; w += (size_t)N_NODES * 128;
    float* ebuf = w; w += (size_t)(N_EDGES + N_NODES) * NHEAD;
    float* mbuf = w; w += (size_t)N_NODES * NHEAD;
    float* sbuf = w; w += (size_t)N_NODES * NHEAD;
    u16* xbf = (u16*)w;  w += ((size_t)N_NODES * TDIM + 1) / 2;   // bf16 stage of x
    u16* hbf = (u16*)w;  w += ((size_t)N_NODES * 128 + 1) / 2;    // bf16 stage of layer input
    u16* wbf = (u16*)w;  w += ((size_t)HC * VDIM + 1) / 2;        // bf16 stage of weights (max 32x2048)
    float* h4   = fx;
    float* tmpv = hA;                                   // NV*HC, pre-layer only
    float* tmpa = hA + (size_t)NVN * HC;
    u16*   vbf  = (u16*)xl;                             // NV*VDIM bf16 (82MB <= xl+xr), pre-layer only
    u16*   abf  = (u16*)accb;                           // NA*ADIM bf16 (20.5MB), pre-layer only

    const int TPB = 256;
    auto cvt = [&](const float* in, u16* out, int n) {
        const int blocks = imin((n / 4 + TPB - 1) / TPB, 8192);
        cvt_bf16_kernel<<<blocks ? blocks : 1, TPB, 0, stream>>>(in, out, n);
    };
    auto gemm = [&](const u16* X, const u16* W, const float* b, float* Y,
                    int n, int K, int C) {
        const int tiles  = (n / 32) * (C / 32);
        const int blocks = imin((tiles + 7) / 8, 4096);
        gemm_bias_wmma_bf16<<<blocks ? blocks : 1, TPB, 0, stream>>>(X, W, b, Y, n, K, C);
    };

    // ---- input projections (bf16 WMMA) + modal fusion ----
    cvt(x, xbf, N_NODES * TDIM);
    cvt(Wt, wbf, HC * TDIM);
    gemm(xbf, wbf, bt, fx, N_NODES, TDIM, HC);
    cvt(x_vision, vbf, NVN * VDIM);
    cvt(Wv, wbf, HC * VDIM);
    gemm(vbf, wbf, bv, tmpv, NVN, VDIM, HC);
    cvt(x_audio, abf, NAN_ * ADIM);
    cvt(Wa, wbf, HC * ADIM);
    gemm(abf, wbf, ba, tmpa, NAN_, ADIM, HC);
    fuse_modal_kernel<<<2500, TPB, 0, stream>>>(fx, x, tmpv, vidx, gv, bev, NVN,  TDIM);
    fuse_modal_kernel<<<2500, TPB, 0, stream>>>(fx, x, tmpa, aidx, ga, bea, NAN_, TDIM);
    cvt(fx, hbf, N_NODES * HC);   // bf16 stage of layer-1 input

    // ---- 4 GATv2 layers ----
    int Din = HC;
    float*       houts[4] = {hA, hB, hA, h4};
    const float* resid[4] = {nullptr, hA, hB, nullptr};
    const int    hs[4]    = {NHEAD, NHEAD, NHEAD, 1};
    for (int L = 0; L < 4; ++L) {
        const int h = hs[L];
        const int D = h * HC;
        cvt(cWl[L], wbf, D * Din);
        gemm(hbf, wbf, cbl[L], xl, N_NODES, Din, D);
        cvt(cWr[L], wbf, D * Din);
        gemm(hbf, wbf, cbr[L], xr, N_NODES, Din, D);
        fill_kernel<<<512,  TPB, 0, stream>>>(mbuf, -INFINITY, N_NODES * h);
        fill_kernel<<<512,  TPB, 0, stream>>>(sbuf, 0.f,       N_NODES * h);
        fill_kernel<<<2048, TPB, 0, stream>>>(accb, 0.f,       N_NODES * D);
        const int total   = (N_EDGES + N_NODES) * h;
        const int eblocks = imin((total + TPB - 1) / TPB, 32768);
        edge_score_kernel<<<eblocks, TPB, 0, stream>>>(xl, xr, catt[L], src, dst,
                                                       ebuf, mbuf, N_EDGES, N_NODES, h);
        edge_softmax_kernel<<<eblocks, TPB, 0, stream>>>(dst, mbuf, ebuf, sbuf,
                                                         N_EDGES, N_NODES, h);
        edge_aggregate_kernel<<<eblocks, TPB, 0, stream>>>(xl, src, dst, ebuf, sbuf,
                                                           accb, N_EDGES, N_NODES, h);
        post_kernel<<<4096, TPB, 0, stream>>>(accb, resid[L], cbias[L], ng[L], nb[L],
                                              houts[L], (L < 3) ? hbf : nullptr,
                                              N_NODES, D, (L < 3) ? 1 : 0);
        Din = D;
    }

    // ---- readout head ----
    head_kernel<<<1, 64, 0, stream>>>(h4, ptr, Wo, bo, (float*)d_out, BATCH);
}
